// HippoRNN_65171833749538
// MI455X (gfx1250) — matrix operand order repaired and checked by reading, not compiled, not run on hardware
//
#include <hip/hip_runtime.h>
#include <hip/hip_bf16.h>
#include <stdint.h>

// Problem constants (from reference): BATCH,T,INPUT,HIDDEN,OUT = 64,512,128,256,10
#define BATCHN 64
#define TLEN   512
#define INPUTN 128
#define HID    256
#define OUTD   10

// ---------------------------------------------------------------------------
// CDNA5 WMMA types (wave32): 16x16x32 bf16 -> f32 accumulate
// ---------------------------------------------------------------------------
typedef __attribute__((ext_vector_type(16))) __bf16 v16bf;
typedef __attribute__((ext_vector_type(8)))  float  v8f;
typedef __attribute__((ext_vector_type(4)))  unsigned int u32x4;
typedef __attribute__((ext_vector_type(4)))  int v4i;
typedef __attribute__((ext_vector_type(8)))  int v8i;

union Frag32B { u32x4 q[2]; v16bf v; };

// A-matrix fragment (16x32 bf16): lanes 0-15 hold K=[0..15],
// lanes 16-31 hold K=[16..31] (contiguous); row M = lane%16. Two 16B chunks.
static __device__ inline v16bf load_a16(const __bf16* rowp, int kk, int hiH) {
  const int k0 = 32 * kk + 16 * hiH;
  Frag32B f;
  f.q[0] = *(const u32x4*)(rowp + k0);
  f.q[1] = *(const u32x4*)(rowp + k0 + 8);
  return f.v;
}

// B-matrix fragment (32x16 bf16): lanes 0-15 hold K=[0..15], lanes 16-31 hold
// K=[16..31] (contiguous 32B); col N = lane%16.
static __device__ inline v16bf load_b16(const __bf16* rowp, int kk, int hiH) {
  const int k0 = 32 * kk + 16 * hiH;
  Frag32B f;
  f.q[0] = *(const u32x4*)(rowp + k0);
  f.q[1] = *(const u32x4*)(rowp + k0 + 8);
  return f.v;
}

// B fragment built from fp32 weights (one-time conversion; loop-invariant).
static __device__ inline v16bf load_b16_from_f32(const float* rowp, int k0) {
  v16bf v;
#pragma unroll
  for (int i = 0; i < 16; ++i) v[i] = (__bf16)rowp[k0 + i];
  return v;
}

static __device__ inline v8f wmma_bf16(v16bf a, v16bf b, v8f c) {
  // v_wmma_f32_16x16x32_bf16 (codegen-confirmed 8-arg form)
  return __builtin_amdgcn_wmma_f32_16x16x32_bf16(false, a, false, b,
                                                 (short)0, c, false, false);
}

// ---------------------------------------------------------------------------
// Phase A: discretized HiPPO-LegS operators.
// M = I - dt/2 A is lower-triangular with rank-1 off-diagonal (A_ij = -q_i q_j),
// so forward substitution reduces to a per-column running sum S_c = Σ q_j X[j][c]:
//   X[i][c] = (N[i][c] - dt/2 * q_i * S_c) / M_ii ,   O(1) per element, no LDS.
// Ad stored as two-term bf16 split (hi + lo) for split-precision WMMA.
// One block per timestep; thread c owns column c. Stores are coalesced per row i.
// ---------------------------------------------------------------------------
__global__ __launch_bounds__(HID) void legs_disc_kernel(
    __bf16* __restrict__ Adh, __bf16* __restrict__ Adl, float* __restrict__ Bd) {
  const int t = blockIdx.x;
  const int c = threadIdx.x;
  const float dt  = 1.0f / (float)(t + 1);
  const float dt2 = 0.5f * dt;
  const float qc  = sqrtf(2.0f * (float)c + 1.0f);

  if (c == 0) {  // Bd: same recurrence, single running sum (scalar, cheap)
    float SB = 0.0f;
    for (int i = 0; i < HID; ++i) {
      const float qi = sqrtf(2.0f * (float)i + 1.0f);
      const float Md = 1.0f + dt2 * (float)(i + 1);
      const float x  = (dt * qi - dt2 * qi * SB) / Md;
      Bd[(size_t)t * HID + i] = x;
      SB += qi * x;
    }
  }

  float Sc = 0.0f;
  for (int i = 0; i < HID; ++i) {
    const float qi = sqrtf(2.0f * (float)i + 1.0f);
    const float Md = 1.0f + dt2 * (float)(i + 1);
    float x = 0.0f;
    if (i >= c) {
      const float Nv = (i == c) ? (1.0f - dt2 * (float)(i + 1)) : (-dt2 * qi * qc);
      x = (Nv - dt2 * qi * Sc) / Md;
      Sc += qi * x;
    }
    const size_t off = ((size_t)t * HID + i) * HID + c;
    const __bf16 xh = (__bf16)x;
    Adh[off] = xh;
    Adl[off] = (__bf16)(x - (float)xh);
  }
}

// ---------------------------------------------------------------------------
// Phase B: hoisted input projections (fully parallel batched WMMA GEMM).
//   U1[t][b][j] = xs[b][t]·Wih[j] + bih[j] + bhh[j]
//   Ug[t][b][j] = xs[b][t]·Wgx[j] + bg[j]
// One block per t: stage xs[:,t,:] as bf16 in LDS, then 512 WMMAs per block (64 per wave).
// ---------------------------------------------------------------------------
__global__ __launch_bounds__(256, 2) void input_proj_kernel(
    const float* __restrict__ xs, const float* __restrict__ Wih,
    const float* __restrict__ bih, const float* __restrict__ bhh,
    const float* __restrict__ Wgx, const float* __restrict__ bg,
    float* __restrict__ U1, float* __restrict__ Ug) {
  __shared__ __align__(16) __bf16 x_s[BATCHN][136];  // 136 pad -> 4-dword bank skew
  const int tid = threadIdx.x, t = blockIdx.x;
  const int wv = tid >> 5, lane = tid & 31, ln = lane & 15, hiH = lane >> 4;

  for (int i = tid; i < BATCHN * INPUTN; i += 256) {
    const int b = i >> 7, k = i & 127;
    x_s[b][k] = (__bf16)xs[((size_t)b * TLEN + t) * INPUTN + k];
  }
  __syncthreads();

#pragma unroll
  for (int nsel = 0; nsel < 2; ++nsel) {
    const int nt = wv + 8 * nsel;
    const int col = 16 * nt + ln;
    v16bf B1[4], B2[4];
#pragma unroll
    for (int kk = 0; kk < 4; ++kk) {
      B1[kk] = load_b16_from_f32(Wih + (size_t)col * INPUTN, 32 * kk + 16 * hiH);
      B2[kk] = load_b16_from_f32(Wgx + (size_t)col * INPUTN, 32 * kk + 16 * hiH);
    }
    const float bv1 = bih[col] + bhh[col];
    const float bv2 = bg[col];
#pragma unroll
    for (int m = 0; m < 4; ++m) {
      v8f a1, a2;
#pragma unroll
      for (int r = 0; r < 8; ++r) { a1[r] = bv1; a2[r] = bv2; }
#pragma unroll
      for (int kk = 0; kk < 4; ++kk) {
        const v16bf A = load_a16(&x_s[16 * m + ln][0], kk, hiH);
        a1 = wmma_bf16(A, B1[kk], a1);
        a2 = wmma_bf16(A, B2[kk], a2);
      }
#pragma unroll
      for (int r = 0; r < 8; ++r) {
        const size_t row = (size_t)t * BATCHN + 16 * m + r + 8 * hiH;
        U1[row * HID + col] = a1[r];
        Ug[row * HID + col] = a2[r];
      }
    }
  }
}

// ---------------------------------------------------------------------------
// Phase C: persistent sequential scan (batch split across 4 WGPs; 16 wave32s).
//  - Whh/Wgh B-fragments register-resident (zero weight traffic in the loop).
//  - Ad_hi[t] (128KB) staged into LDS by the Tensor Data Mover: one
//    tensor_load_to_lds per step (wave 0), TENSORcnt-tracked, overlapped with
//    phase-1 compute. D# pad (interval=128 dwords, amount=4 dwords) deposits
//    rows at a 528B stride -> conflict-free ds_load_b128 fragment reads.
//  - c recurrence: two-term bf16 split (c_hi+c_lo) x (Ad_hi+Ad_lo), 3 WMMAs
//    per k-tile (lo*lo dropped) -> ~fp32-ish accuracy at bf16 WMMA speed.
//  - U1/Ug tiles are single-use per WG -> non-temporal loads.
//  - 2 workgroup barriers per step; f = h·Wf + bf via shuffles + ds_add_f32.
// ~203KB LDS also pins one workgroup per WGP.
// ---------------------------------------------------------------------------
__global__ __launch_bounds__(512, 1) void hippo_scan_kernel(
    const float* __restrict__ Whh, const float* __restrict__ Wgh,
    const float* __restrict__ Wf,  const float* __restrict__ bfp,
    const float* __restrict__ Wfc, const float* __restrict__ bfc,
    const float* __restrict__ U1,  const float* __restrict__ Ug,
    const __bf16* __restrict__ Adh, const __bf16* __restrict__ Adl,
    const float* __restrict__ Bd, float* __restrict__ out) {
  // 264-elem row stride (528B = 132 dwords, %64 == 4) -> conflict-free frag loads
  __shared__ __align__(16) __bf16 h_s  [2][16][264];
  __shared__ __align__(16) __bf16 chi_s[2][16][264];
  __shared__ __align__(16) __bf16 clo_s[2][16][264];
  __shared__ __align__(16) __bf16 adh_s[HID][264];   // TDM-padded Ad_hi[t] tile
  __shared__ float f_s[2][16];
  __shared__ float c32_s[16][260];

  const int tid = threadIdx.x;
  const int w   = tid >> 5;     // wave id = owned column tile
  const int lane = tid & 31;
  const int ln  = lane & 15;
  const int hiH = lane >> 4;
  const int g   = blockIdx.x;   // batch tile (16 rows)
  const int col = 16 * w + ln;

  // Loop-invariant weight fragments -> VGPRs (8 k-tiles x 2 GEMMs = 128 VGPRs)
  v16bf Bh[8], Bg2[8];
#pragma unroll
  for (int kk = 0; kk < 8; ++kk) {
    Bh[kk]  = load_b16_from_f32(Whh + (size_t)col * HID, 32 * kk + 16 * hiH);
    Bg2[kk] = load_b16_from_f32(Wgh + (size_t)col * HID, 32 * kk + 16 * hiH);
  }
  const float wfv = Wf[col];
  const float bf0 = bfp[0];

  float hreg[8];  // fp32 h, D-fragment layout: row = r + 8*hiH, col = col
#pragma unroll
  for (int r = 0; r < 8; ++r) hreg[r] = 0.0f;

  for (int i = tid; i < 2 * 16 * 264; i += 512) {
    ((__bf16*)h_s)[i]   = (__bf16)0.0f;
    ((__bf16*)chi_s)[i] = (__bf16)0.0f;
    ((__bf16*)clo_s)[i] = (__bf16)0.0f;
  }
  if (tid < 32) ((float*)f_s)[tid] = 0.0f;
  const unsigned int adh_lds_off = (unsigned int)(uintptr_t)(&adh_s[0][0]);
  __syncthreads();

  for (int t = 0; t < TLEN; ++t) {
    const int p = t & 1, np = p ^ 1;

    // ---- TDM: DMA Ad_hi[t] (128KB) -> LDS, overlapped with phase 1 --------
    // Previous readers of adh_s finished before the trailing barrier of step
    // t-1, so the buffer is free here. Wave 0 issues; completion is published
    // to the other waves by wait_tensorcnt + the phase-1 barrier.
    if (w == 0) {
      const unsigned long long ga =
          (unsigned long long)(uintptr_t)Adh + (unsigned long long)t * (HID * HID * 2);
      v4i g0; v8i g1; v4i g2; v4i g3;
      g0[0] = 1;                                   // count=1, load descriptor
      g0[1] = (int)adh_lds_off;                    // LDS dest (bytes)
      g0[2] = (int)(unsigned int)ga;               // global addr [31:0]
      g0[3] = (int)(((unsigned int)(ga >> 32) & 0x01FFFFFFu) | 0x80000000u); // [56:32] | type=2
      g1[0] = (2 << 16)        // data_size = 4B units
            | (1 << 20)        // pad_enable
            | (6 << 22)        // pad_interval: every 128 dwords (= one 512B row)
            | (3 << 25);       // pad_amount: 4 dwords -> 528B LDS row stride
      g1[1] = (int)0x80000000u;                    // tensor_dim0 = 32768 dwords (lo16<<16)
      g1[2] = (1 << 16);                           // tensor_dim0 hi=0; tensor_dim1 = 1
      g1[3] = (int)0x80000000u;                    // tile_dim0 = 32768 dwords
      g1[4] = 0;                                   // tile_dim1/2 unused
      g1[5] = 32768;                               // tensor_dim0_stride (dwords)
      g1[6] = 0;
      g1[7] = 0;
      g2 = (v4i){0, 0, 0, 0};
      g3 = (v4i){0, 0, 0, 0};
      asm volatile("tensor_load_to_lds %0, %1, %2, %3"
                   :: "s"(g0), "s"(g1), "s"(g2), "s"(g3)
                   : "memory");
    }

    // ---- phase 1: pre-activations, gating, f partials -------------------
    const float* u1 = U1 + (((size_t)t * BATCHN + 16 * g + 8 * hiH) * HID) + col;
    const float* ug = Ug + (((size_t)t * BATCHN + 16 * g + 8 * hiH) * HID) + col;
    v8f a1, ag;
#pragma unroll
    for (int r = 0; r < 8; ++r) {  // single-use streams -> non-temporal
      a1[r] = __builtin_nontemporal_load(u1 + (size_t)r * HID);
      ag[r] = __builtin_nontemporal_load(ug + (size_t)r * HID);
    }
#pragma unroll
    for (int kk = 0; kk < 8; ++kk) {
      const v16bf A = load_a16(&h_s[p][ln][0], kk, hiH);
      a1 = wmma_bf16(A, Bh[kk],  a1);
      ag = wmma_bf16(A, Bg2[kk], ag);
    }
#pragma unroll
    for (int r = 0; r < 8; ++r) {
      const float x1 = fminf(fmaxf(a1[r], -20.0f), 20.0f);
      const float e2 = __expf(2.0f * x1);
      const float ht = (e2 - 1.0f) / (e2 + 1.0f);              // tanh
      const float gg = 1.0f / (1.0f + __expf(-ag[r]));         // sigmoid
      const float hn = hreg[r] + gg * (ht - hreg[r]);          // (1-g)h + g*h~
      hreg[r] = hn;
      h_s[np][r + 8 * hiH][col] = (__bf16)hn;
      float pr = hn * wfv;                                      // f partial
#pragma unroll
      for (int m = 1; m < 16; m <<= 1) pr += __shfl_xor(pr, m, 16);
      if (ln == 0) atomicAdd(&f_s[p][r + 8 * hiH], pr);         // ds_add_f32
    }
    if (tid < 16) f_s[np][tid] = 0.0f;  // dead buffer -> next step accumulator
    if (w == 0) __builtin_amdgcn_s_wait_tensorcnt(0);  // Ad_hi[t] landed in LDS
    __syncthreads();

    // ---- phase 2: c = c·Ad^T + (f+bf)·Bd  (split-bf16 WMMA) -------------
    const __bf16* adh_row = &adh_s[col][0];                   // LDS (TDM tile)
    const __bf16* adl = Adl + ((size_t)t * HID + col) * HID;  // global (L2-shared)
    const float bd = Bd[(size_t)t * HID + col];
    v8f cacc;
#pragma unroll
    for (int r = 0; r < 8; ++r) cacc[r] = (f_s[p][r + 8 * hiH] + bf0) * bd;
#pragma unroll
    for (int kk = 0; kk < 8; ++kk) {
      const v16bf Ah  = load_a16(&chi_s[p][ln][0], kk, hiH);
      const v16bf Al  = load_a16(&clo_s[p][ln][0], kk, hiH);
      const v16bf Bhk = load_b16(adh_row, kk, hiH);
      const v16bf Blk = load_b16(adl, kk, hiH);
      cacc = wmma_bf16(Ah, Bhk, cacc);   // hi*hi
      cacc = wmma_bf16(Al, Bhk, cacc);   // lo*hi
      cacc = wmma_bf16(Ah, Blk, cacc);   // hi*lo   (lo*lo dropped: ~2^-17)
    }
#pragma unroll
    for (int r = 0; r < 8; ++r) {
      const float cv = cacc[r];
      const __bf16 ch = (__bf16)cv;
      chi_s[np][r + 8 * hiH][col] = ch;
      clo_s[np][r + 8 * hiH][col] = (__bf16)(cv - (float)ch);
    }
    if (t == TLEN - 1) {
#pragma unroll
      for (int r = 0; r < 8; ++r) c32_s[r + 8 * hiH][col] = cacc[r];
    }
    if (t + 1 < TLEN) {  // warm L2 for next step's streams (global_prefetch_b8)
      __builtin_prefetch(adl + (size_t)HID * HID, 0, 1);
      __builtin_prefetch(u1 + (size_t)BATCHN * HID, 0, 1);
      __builtin_prefetch(ug + (size_t)BATCHN * HID, 0, 1);
    }
    __syncthreads();
  }

  // ---- epilogue: out = c · Wfc^T + bfc (16 rows x 10 outs per WG) -------
  if (tid < 16 * OUTD) {
    const int lr = tid / OUTD, o = tid % OUTD;
    float s = bfc[o];
    for (int j = 0; j < HID; ++j) s += c32_s[lr][j] * Wfc[o * HID + j];
    out[(16 * g + lr) * OUTD + o] = s;
  }
}

// ---------------------------------------------------------------------------
// Launcher. Workspace layout (≈202 MB):
//   Adh (bf16 T*H*H) | Adl (bf16 T*H*H) | Bd (f32 T*H) | U1 (f32 T*B*H) | Ug
// ---------------------------------------------------------------------------
extern "C" void kernel_launch(void* const* d_in, const int* in_sizes, int n_in,
                              void* d_out, int out_size, void* d_ws, size_t ws_size,
                              hipStream_t stream) {
  const float* xs  = (const float*)d_in[0];
  const float* Wih = (const float*)d_in[1];
  const float* bih = (const float*)d_in[2];
  const float* Whh = (const float*)d_in[3];
  const float* bhh = (const float*)d_in[4];
  const float* Wgx = (const float*)d_in[5];
  const float* Wgh = (const float*)d_in[6];
  const float* bg  = (const float*)d_in[7];
  const float* Wf  = (const float*)d_in[8];
  const float* bf  = (const float*)d_in[9];
  const float* Wfc = (const float*)d_in[10];
  const float* bfc = (const float*)d_in[11];

  char* ws = (char*)d_ws;
  __bf16* Adh = (__bf16*)ws; ws += (size_t)TLEN * HID * HID * sizeof(__bf16);
  __bf16* Adl = (__bf16*)ws; ws += (size_t)TLEN * HID * HID * sizeof(__bf16);
  float*  Bd  = (float*)ws;  ws += (size_t)TLEN * HID * sizeof(float);
  float*  U1  = (float*)ws;  ws += (size_t)TLEN * BATCHN * HID * sizeof(float);
  float*  Ug  = (float*)ws;  ws += (size_t)TLEN * BATCHN * HID * sizeof(float);

  legs_disc_kernel<<<dim3(TLEN), dim3(HID), 0, stream>>>(Adh, Adl, Bd);
  input_proj_kernel<<<dim3(TLEN), dim3(256), 0, stream>>>(xs, Wih, bih, bhh,
                                                          Wgx, bg, U1, Ug);
  hippo_scan_kernel<<<dim3(4), dim3(512), 0, stream>>>(Whh, Wgh, Wf, bf, Wfc, bfc,
                                                       U1, Ug, Adh, Adl, Bd,
                                                       (float*)d_out);
}
